// TAUS_67284957659496
// MI455X (gfx1250) — matrix operand order repaired and checked by reading, not compiled, not run on hardware
//
#include <hip/hip_runtime.h>
#include <hip/hip_bf16.h>

typedef __attribute__((ext_vector_type(16))) __bf16 v16bf;
typedef __attribute__((ext_vector_type(8)))  __bf16 v8bf;
typedef __attribute__((ext_vector_type(8)))  float  v8f;

#define H_IN   48
#define W_IN   48
#define H_OUT  192
#define W_OUT  192
#define IN_C   32
#define OUT_C  3
#define FT_DIM 16
#define NPIX   (2 * H_OUT * W_OUT)

// per-wave LDS layout (bytes)
#define OFF_SEL   0          // [16][512] bf16 = 16384
#define OFF_U     16384      // [16][768] bf16 = 24576
#define OFF_G     40960      // [16][384] bf16 = 12288
#define OFF_T     53248      // [16][192] bf16 = 6144
#define OFF_F1    59392      // [16][192] bf16 = 6144
#define OFF_OSF   65536      // [16][32]  bf16 = 1024
#define OFF_GEO   66560      // [16][8]   int  = 512
#define WSTR      67072
#define WAVES_PER_BLOCK 4

__device__ __forceinline__ int refl48(int v) {
    v = v < 0 ? -v : v;
    return v >= 48 ? 94 - v : v;
}
__device__ __forceinline__ __bf16 f2bf(float f) { return (__bf16)f; }
__device__ __forceinline__ float  bf2f(__bf16 h) { return (float)h; }

// CDNA5 has a hardware V_TANH_F32 transcendental; use it when the builtin exists.
__device__ __forceinline__ float fast_tanh(float x) {
#if __has_builtin(__builtin_amdgcn_tanhf)
    return __builtin_amdgcn_tanhf(x);
#elif __has_builtin(__builtin_amdgcn_tanh_f32)
    return __builtin_amdgcn_tanh_f32(x);
#else
    return tanhf(x);
#endif
}

// ---------------------------------------------------------------------------
// Stage 1a: feature_trans 4x4 conv with reflect pad (1,2,1,2)
// ---------------------------------------------------------------------------
__global__ void TAUS_ft_conv(const float* __restrict__ lr,
                             const float* __restrict__ ftw,
                             const float* __restrict__ ftb,
                             float* __restrict__ ft) {
    int idx = blockIdx.x * blockDim.x + threadIdx.x;
    if (idx >= 2 * FT_DIM * H_IN * W_IN) return;
    int x  = idx % W_IN;
    int y  = (idx / W_IN) % H_IN;
    int oc = (idx / (W_IN * H_IN)) % FT_DIM;
    int b  = idx / (W_IN * H_IN * FT_DIM);
    float acc = ftb[oc];
    for (int c = 0; c < IN_C; ++c) {
        const float* src = lr + ((size_t)(b * IN_C + c)) * H_IN * W_IN;
        const float* wk  = ftw + ((size_t)(oc * IN_C + c)) * 16;
        for (int kh = 0; kh < 4; ++kh) {
            int ry = refl48(y + kh - 1);
            for (int kw = 0; kw < 4; ++kw) {
                int rx = refl48(x + kw - 1);
                acc = fmaf(src[ry * W_IN + rx], wk[kh * 4 + kw], acc);
            }
        }
    }
    ft[idx] = acc;
}

// ---------------------------------------------------------------------------
// Stage 1b: weight fp32 [K][N] -> bf16 [N][Kpad] (transposed, K zero-padded)
// ---------------------------------------------------------------------------
__global__ void TAUS_convw(const float* __restrict__ src, __bf16* __restrict__ dst,
                           int K, int N, int Kpad) {
    int idx = blockIdx.x * blockDim.x + threadIdx.x;
    if (idx >= N * Kpad) return;
    int kp = idx % Kpad, n = idx / Kpad;
    float v = (kp < K) ? src[(size_t)kp * N + n] : 0.0f;
    dst[idx] = f2bf(v);
}

// ---------------------------------------------------------------------------
// WMMA fragment helpers (CDNA5 16-bit A/B layouts, wave32)
// ---------------------------------------------------------------------------
__device__ __forceinline__ v8f wmma_bf16(v16bf a, v16bf b, v8f c) {
    return __builtin_amdgcn_wmma_f32_16x16x32_bf16(false, a, false, b,
                                                   (short)0, c, false, false);
}

// A: LDS row-major [16][Kpad] bf16. lane m=lane&15; elems 0..7 -> K=koff..koff+7,
// elems 8..15 -> K=16+koff..23+koff, koff = (lane>=16)?8:0
__device__ __forceinline__ v16bf load_a_frag(const __bf16* act, int Kpad,
                                             int ktile, int lane) {
    int m    = lane & 15;
    int koff = (lane >> 4) << 3;
    int base = m * Kpad + ktile * 32 + koff;
    v8bf lo = *(const v8bf*)(act + base);
    v8bf hi = *(const v8bf*)(act + base + 16);
    v16bf a;
#pragma unroll
    for (int e = 0; e < 8; ++e) { a[e] = lo[e]; a[e + 8] = hi[e]; }
    return a;
}

// B: global N-major [N][Kpad] bf16. lane col=lane&15; elems 0..7 -> K = koff2+e,
// elems 8..15 -> K = 16+koff2+e, koff2 = (lane>>=16)?8:0 -> 2x b128
__device__ __forceinline__ v16bf load_b_frag(const __bf16* w, int Kpad,
                                             int ktile, int ntile, int lane) {
    int col   = ntile * 16 + (lane & 15);
    int koff2 = (lane >> 4) << 3;
    const v8bf* p = (const v8bf*)(w + (size_t)col * Kpad + ktile * 32 + koff2);
    v8bf lo = p[0], hi = p[2];
    v16bf b;
#pragma unroll
    for (int e = 0; e < 8; ++e) { b[e] = lo[e]; b[e + 8] = hi[e]; }
    return b;
}

// D: (vgpr r, lane l) -> row p = r + 8*(l>>4), col n = ntile*16 + (l&15)
__device__ __forceinline__ void store_d(__bf16* act, int Kpad, int ntile,
                                        int lane, v8f d) {
    int n  = ntile * 16 + (lane & 15);
    int pb = (lane >> 4) << 3;
#pragma unroll
    for (int r = 0; r < 8; ++r) act[(pb + r) * Kpad + n] = f2bf(d[r]);
}

// ---------------------------------------------------------------------------
// Stage 2: fused geometry + gather + per-pixel MLP (WMMA) + slice_mul + tanh
// One wave32 per 16-pixel M-tile, 4 waves per block, N-tiles blocked by 4.
// ---------------------------------------------------------------------------
__global__ void TAUS_main(
    const float* __restrict__ lr, const float* __restrict__ ft,
    const __bf16* __restrict__ W1A, const __bf16* __restrict__ W1B,
    const __bf16* __restrict__ W2A, const __bf16* __restrict__ W2B,
    const __bf16* __restrict__ W3A, const __bf16* __restrict__ W3B,
    const float* __restrict__ l1b1, const float* __restrict__ l1b2,
    const float* __restrict__ l2b1, const float* __restrict__ l2b2,
    const float* __restrict__ l3b1, const float* __restrict__ l3b2,
    float* __restrict__ out)
{
    extern __shared__ char smem[];
    const int lane = threadIdx.x & 31;
    const int wave = threadIdx.x >> 5;
    char* wb = smem + wave * WSTR;
    __bf16* SEL = (__bf16*)(wb + OFF_SEL);
    __bf16* U   = (__bf16*)(wb + OFF_U);
    __bf16* G   = (__bf16*)(wb + OFF_G);
    __bf16* T   = (__bf16*)(wb + OFF_T);
    __bf16* F1  = (__bf16*)(wb + OFF_F1);
    __bf16* OSF = (__bf16*)(wb + OFF_OSF);
    int*    GEO = (int*)   (wb + OFF_GEO);

    const int tile0 = (blockIdx.x * WAVES_PER_BLOCK + wave) * 16;

    // --- geometry + osf vector (one pixel per lane, lanes 0..15)
    if (lane < 16) {
        int pid = tile0 + lane;
        int b   = pid / (H_OUT * W_OUT);
        int rem = pid % (H_OUT * W_OUT);
        int y = rem / W_OUT, x = rem % W_OUT;
        float ch = (y + 1) * 0.25f - 0.625f;
        float cw = (x + 1) * 0.25f - 0.625f;
        float fh = floorf(ch), fw = floorf(cw);
        int ih = (int)fh, iw = (int)fw;
#pragma unroll
        for (int t = 0; t < 4; ++t) {
            GEO[lane * 8 + t]     = refl48(ih + t - 1);
            GEO[lane * 8 + 4 + t] = refl48(iw + t - 1);
        }
        int cy = refl48(ih), cx = refl48(iw);
        const float* fp = ft + (((size_t)b * FT_DIM) * H_IN + cy) * W_IN + cx;
#pragma unroll
        for (int c = 0; c < FT_DIM; ++c)
            OSF[lane * 32 + c] = f2bf(fp[(size_t)c * H_IN * W_IN]);
        OSF[lane * 32 + 16] = f2bf(fh - ch);
        OSF[lane * 32 + 17] = f2bf(fw - cw);
        OSF[lane * 32 + 18] = f2bf(4.0f);
        OSF[lane * 32 + 19] = f2bf(4.0f);
#pragma unroll
        for (int c = 20; c < 32; ++c) OSF[lane * 32 + c] = f2bf(0.0f);
    }
    __syncthreads();

    // --- stage sel[16][512] (reflect-gathered K*K neighborhoods), bf16
    {
        int b0 = tile0 / (H_OUT * W_OUT);   // tiles never straddle batch
        const float* lrb = lr + (size_t)b0 * IN_C * H_IN * W_IN;
        for (int i = lane; i < 16 * 512; i += 32) {
            int m = i >> 9, j = i & 511;
            int c = j >> 4, kh = (j >> 2) & 3, kw = j & 3;
            int ry = GEO[m * 8 + kh], rx = GEO[m * 8 + 4 + kw];
            SEL[m * 512 + j] = f2bf(lrb[((size_t)c * H_IN + ry) * W_IN + rx]);
        }
    }
    __syncthreads();

    // --- branch layer A: t = tanh(osf @ WA + bA)  [16x32]x[32x192]
    auto layerA = [&](const __bf16* WA, const float* bA) {
        v16bf a = load_a_frag(OSF, 32, 0, lane);
#pragma unroll
        for (int ng = 0; ng < 3; ++ng) {
            v8f acc[4] = {{}, {}, {}, {}};
#pragma unroll
            for (int q = 0; q < 4; ++q) {
                v16bf bb = load_b_frag(WA, 32, 0, ng * 4 + q, lane);
                acc[q] = wmma_bf16(a, bb, acc[q]);
            }
#pragma unroll
            for (int q = 0; q < 4; ++q) {
                int nt = ng * 4 + q;
                float bias = bA[nt * 16 + (lane & 15)];
                v8f d;
#pragma unroll
                for (int r = 0; r < 8; ++r) d[r] = fast_tanh(acc[q][r] + bias);
                store_d(T, 192, nt, lane, d);
            }
        }
    };

    // --- branch 1: f1 = t1 @ l1w2 + b   [16x192]x[192x192]
    layerA(W1A, l1b1);
    for (int ng = 0; ng < 3; ++ng) {
        v8f acc[4] = {{}, {}, {}, {}};
        for (int kt = 0; kt < 6; ++kt) {
            v16bf a = load_a_frag(T, 192, kt, lane);
#pragma unroll
            for (int q = 0; q < 4; ++q) {
                v16bf bb = load_b_frag(W1B, 192, kt, ng * 4 + q, lane);
                acc[q] = wmma_bf16(a, bb, acc[q]);
            }
        }
#pragma unroll
        for (int q = 0; q < 4; ++q) {
            int nt = ng * 4 + q;
            float bias = l1b2[nt * 16 + (lane & 15)];
            v8f d;
#pragma unroll
            for (int r = 0; r < 8; ++r) d[r] = acc[q][r] + bias;
            store_d(F1, 192, nt, lane, d);
        }
    }

    // --- branch 2 + merge: g = tanh([f1*f2, f1])  -> G[16][384]
    layerA(W2A, l2b1);
    for (int ng = 0; ng < 3; ++ng) {
        v8f acc[4] = {{}, {}, {}, {}};
        for (int kt = 0; kt < 6; ++kt) {
            v16bf a = load_a_frag(T, 192, kt, lane);
#pragma unroll
            for (int q = 0; q < 4; ++q) {
                v16bf bb = load_b_frag(W2B, 192, kt, ng * 4 + q, lane);
                acc[q] = wmma_bf16(a, bb, acc[q]);
            }
        }
#pragma unroll
        for (int q = 0; q < 4; ++q) {
            int nt = ng * 4 + q;
            float bias = l2b2[nt * 16 + (lane & 15)];
            int n  = nt * 16 + (lane & 15);
            int pb = (lane >> 4) << 3;
#pragma unroll
            for (int r = 0; r < 8; ++r) {
                float f2v = acc[q][r] + bias;
                float f1v = bf2f(F1[(pb + r) * 192 + n]);
                G[(pb + r) * 384 + n]       = f2bf(fast_tanh(f1v * f2v));
                G[(pb + r) * 384 + 192 + n] = f2bf(fast_tanh(f1v));
            }
        }
    }

    // --- u = leaky(g @ l3w1 + b)  [16x384]x[384x768]
    for (int ng = 0; ng < 12; ++ng) {
        if (ng + 1 < 12)
            __builtin_prefetch(W3A + (size_t)(ng + 1) * 4 * 16 * 384, 0, 1);
        v8f acc[4] = {{}, {}, {}, {}};
        for (int kt = 0; kt < 12; ++kt) {
            v16bf a = load_a_frag(G, 384, kt, lane);
#pragma unroll
            for (int q = 0; q < 4; ++q) {
                v16bf bb = load_b_frag(W3A, 384, kt, ng * 4 + q, lane);
                acc[q] = wmma_bf16(a, bb, acc[q]);
            }
        }
#pragma unroll
        for (int q = 0; q < 4; ++q) {
            int nt = ng * 4 + q;
            float bias = l3b1[nt * 16 + (lane & 15)];
            v8f d;
#pragma unroll
            for (int r = 0; r < 8; ++r) {
                float v = acc[q][r] + bias;
                d[r] = v > 0.0f ? v : 0.01f * v;
            }
            store_d(U, 768, nt, lane, d);
        }
    }

    // --- w = u @ l3w2 + b  [16x768]x[768x1536], fused with sel contraction
    float oacc[OUT_C][8];
#pragma unroll
    for (int o = 0; o < OUT_C; ++o)
#pragma unroll
        for (int r = 0; r < 8; ++r) oacc[o][r] = 0.0f;

    for (int ng = 0; ng < 24; ++ng) {
        if (ng + 1 < 24)
            __builtin_prefetch(W3B + (size_t)(ng + 1) * 4 * 16 * 768, 0, 1);
        v8f acc[4] = {{}, {}, {}, {}};
        for (int kt = 0; kt < 24; ++kt) {
            v16bf a = load_a_frag(U, 768, kt, lane);
#pragma unroll
            for (int q = 0; q < 4; ++q) {
                v16bf bb = load_b_frag(W3B, 768, kt, ng * 4 + q, lane);
                acc[q] = wmma_bf16(a, bb, acc[q]);
            }
        }
#pragma unroll
        for (int q = 0; q < 4; ++q) {
            int nt  = ng * 4 + q;          // groups of 4 never straddle o
            int col = nt * 16 + (lane & 15);
            float bias = l3b2[col];
            int o  = nt >> 5;              // 32 ntiles per output channel
            int j  = col & 511;
            int pb = (lane >> 4) << 3;
#pragma unroll
            for (int r = 0; r < 8; ++r) {
                float wv = acc[q][r] + bias;
                float sv = bf2f(SEL[(pb + r) * 512 + j]);
                oacc[o][r] = fmaf(wv, sv, oacc[o][r]);
            }
        }
    }

    // reduce over the 16-lane N dimension (lanes 0..15 share row r, 16..31 row r+8)
#pragma unroll
    for (int o = 0; o < OUT_C; ++o)
#pragma unroll
        for (int r = 0; r < 8; ++r) {
            float v = oacc[o][r];
            v += __shfl_xor(v, 1, 16);
            v += __shfl_xor(v, 2, 16);
            v += __shfl_xor(v, 4, 16);
            v += __shfl_xor(v, 8, 16);
            oacc[o][r] = v;
        }

    if ((lane & 15) == 0) {
        int pb = (lane >> 4) << 3;
#pragma unroll
        for (int r = 0; r < 8; ++r) {
            int pid = tile0 + pb + r;
            int b   = pid / (H_OUT * W_OUT);
            int rem = pid % (H_OUT * W_OUT);
#pragma unroll
            for (int o = 0; o < OUT_C; ++o)
                out[((size_t)(b * OUT_C + o)) * H_OUT * W_OUT + rem] = fast_tanh(oacc[o][r]);
        }
    }
}

// ---------------------------------------------------------------------------
// Launch
// ---------------------------------------------------------------------------
extern "C" void kernel_launch(void* const* d_in, const int* in_sizes, int n_in,
                              void* d_out, int out_size, void* d_ws, size_t ws_size,
                              hipStream_t stream) {
    const float* lr   = (const float*)d_in[0];
    // d_in[1] kernel_pc unused by reference; d_in[2]/d_in[3] h_out/w_out static
    const float* ftw  = (const float*)d_in[4];
    const float* ftb  = (const float*)d_in[5];
    const float* l1w1 = (const float*)d_in[6];
    const float* l1b1 = (const float*)d_in[7];
    const float* l1w2 = (const float*)d_in[8];
    const float* l1b2 = (const float*)d_in[9];
    const float* l2w1 = (const float*)d_in[10];
    const float* l2b1 = (const float*)d_in[11];
    const float* l2w2 = (const float*)d_in[12];
    const float* l2b2 = (const float*)d_in[13];
    const float* l3w1 = (const float*)d_in[14];
    const float* l3b1 = (const float*)d_in[15];
    const float* l3w2 = (const float*)d_in[16];
    const float* l3b2 = (const float*)d_in[17];
    float* out = (float*)d_out;

    char* ws = (char*)d_ws;
    float*  ftbuf = (float*)(ws + 0);                 // 294912 B
    __bf16* W1A = (__bf16*)(ws + 294912);             // 192*32
    __bf16* W1B = (__bf16*)(ws + 307200);             // 192*192
    __bf16* W2A = (__bf16*)(ws + 380928);
    __bf16* W2B = (__bf16*)(ws + 393216);
    __bf16* W3A = (__bf16*)(ws + 466944);             // 768*384
    __bf16* W3B = (__bf16*)(ws + 1056768);            // 1536*768

    // Stage 1a: ft conv
    {
        int total = 2 * FT_DIM * H_IN * W_IN;
        TAUS_ft_conv<<<(total + 255) / 256, 256, 0, stream>>>(lr, ftw, ftb, ftbuf);
    }
    // Stage 1b: weight conversion (transpose to N-major bf16, K padded)
    TAUS_convw<<<(192 * 32   + 255) / 256, 256, 0, stream>>>(l1w1, W1A, 20,  192,  32);
    TAUS_convw<<<(192 * 192  + 255) / 256, 256, 0, stream>>>(l1w2, W1B, 192, 192,  192);
    TAUS_convw<<<(192 * 32   + 255) / 256, 256, 0, stream>>>(l2w1, W2A, 20,  192,  32);
    TAUS_convw<<<(192 * 192  + 255) / 256, 256, 0, stream>>>(l2w2, W2B, 192, 192,  192);
    TAUS_convw<<<(768 * 384  + 255) / 256, 256, 0, stream>>>(l3w1, W3A, 384, 768,  384);
    TAUS_convw<<<(1536 * 768 + 255) / 256, 256, 0, stream>>>(l3w2, W3B, 768, 1536, 768);

    // Stage 2: fused WMMA MLP + dynamic-filter application
    {
        int blocks = NPIX / (WAVES_PER_BLOCK * 16);   // 1152
        size_t shmem = (size_t)WAVES_PER_BLOCK * WSTR; // 268288 B < 320 KB/WGP
        TAUS_main<<<blocks, WAVES_PER_BLOCK * 32, shmem, stream>>>(
            lr, ftbuf, W1A, W1B, W2A, W2B, W3A, W3B,
            l1b1, l1b2, l2b1, l2b2, l3b1, l3b2, out);
    }
}
